// RNN_35364760715347
// MI455X (gfx1250) — matrix operand order repaired and checked by reading, not compiled, not run on hardware
//
#include <hip/hip_runtime.h>

// Problem sizes
#define B_SZ    64
#define T_SZ    512
#define IN0     256
#define HID     512
#define NT      32          // HID/16 n-tiles
#define LDS_PAD 8           // bf16 elements of row padding (bank-conflict avoidance)
#define KT_ALL  16          // HID/32 k-tiles in the recurrence
#define KT_LDS  8           // k-tiles of W_hh resident in LDS (256 KB)

typedef __attribute__((ext_vector_type(16))) __bf16 v16bf;
typedef __attribute__((ext_vector_type(8)))  float  v8f;

union FragB { v16bf bf; unsigned int w[8]; };
union FragC { v8f f; float x[8]; };

__device__ __forceinline__ unsigned short f2bf(float f) {
    unsigned int u = __builtin_bit_cast(unsigned int, f);
    u += 0x7FFFu + ((u >> 16) & 1u);       // round-to-nearest-even
    return (unsigned short)(u >> 16);
}

__device__ __forceinline__ FragB ld_frag_g(const uint4* p) {
    FragB b;
    uint4 b0 = p[0], b1 = p[1];
    b.w[0] = b0.x; b.w[1] = b0.y; b.w[2] = b0.z; b.w[3] = b0.w;
    b.w[4] = b1.x; b.w[5] = b1.y; b.w[6] = b1.z; b.w[7] = b1.w;
    return b;
}

// ---------------------------------------------------------------------------
// Pack fp32 W[H][K] (row-major) into bf16 WMMA B-fragments.
// Fragment (kt, nt) covers B[k][n] = W[n_global][k_global]; lane L holds
// n = L&15, k = kt*32 + (L<16?0:8) + i + (i<8?0:8), 16 bf16 (32 B) at
// frag*1024 + L*32.
// ---------------------------------------------------------------------------
__global__ void pack_w_kernel(const float* __restrict__ W,
                              unsigned int* __restrict__ dst, int K) {
    int gid  = blockIdx.x * blockDim.x + threadIdx.x;
    int frag = gid >> 5;
    int lane = gid & 31;
    int KT   = K >> 5;
    if (frag >= KT * NT) return;
    int kt = frag >> 5;          // frag = kt*NT + nt, NT == 32
    int nt = frag & 31;
    int j  = nt * 16 + (lane & 15);
    int kb = kt * 32 + ((lane & 16) ? 8 : 0);
    const float* src = W + (size_t)j * K + kb;
    unsigned int out[8];
#pragma unroll
    for (int p = 0; p < 8; ++p) {
        unsigned int lo = f2bf(src[2 * p + (p >= 4 ? 8 : 0)]);
        unsigned int hi = f2bf(src[2 * p + 1 + (p >= 4 ? 8 : 0)]);
        out[p] = lo | (hi << 16);
    }
    uint4* d4 = (uint4*)(dst + (size_t)frag * 256 + lane * 8);
    d4[0] = make_uint4(out[0], out[1], out[2], out[3]);
    d4[1] = make_uint4(out[4], out[5], out[6], out[7]);
}

// ---------------------------------------------------------------------------
// Projection GEMM (templated K, input mode). Grid: T*4 blocks / 256 threads.
// Stores raw C-fragments: proj[((t*32+nt)*4+slice)*256 + lane*8 .. +7].
// MODE 0: rows from fp32 x[B][T][K].  MODE 1: rows from packed bf16 h-frags.
// ---------------------------------------------------------------------------
template <int K, int MODE>
__global__ void __launch_bounds__(256) proj_kernel(
    const void* __restrict__ in,
    const unsigned int* __restrict__ Wp,
    const float* __restrict__ bi1, const float* __restrict__ bi2,
    float* __restrict__ proj)
{
    __shared__ unsigned short hbuf[16 * (HID + LDS_PAD)];
    constexpr int LDSTR = K + LDS_PAD;
    constexpr int KT    = K >> 5;
    const int t     = blockIdx.x >> 2;
    const int slice = blockIdx.x & 3;
    const int tid   = threadIdx.x;
    const int w     = tid >> 5;
    const int lane  = tid & 31;

    if (MODE == 0) {
        const float* x = (const float*)in;
        int m   = tid >> 4;                 // 16 threads per row
        constexpr int cnt = K / 16;         // elems per thread
        int k0  = (tid & 15) * cnt;
        const float* src = x + ((size_t)(slice * 16 + m) * T_SZ + t) * K + k0;
#pragma unroll
        for (int i = 0; i < cnt; i += 2) {
            unsigned int lo = f2bf(src[i]);
            unsigned int hi = f2bf(src[i + 1]);
            *(unsigned int*)&hbuf[m * LDSTR + k0 + i] = lo | (hi << 16);
        }
    } else {
        const uint4* hseq = (const uint4*)in;
        int n = lane & 15;
        for (int nt = w; nt < NT; nt += 8) {
            size_t frag = ((size_t)t * NT + nt) * 4 + slice;
            uint4 v = hseq[frag * 32 + lane];
            unsigned int vv[4] = {v.x, v.y, v.z, v.w};
#pragma unroll
            for (int r = 0; r < 8; ++r) {
                int m = (lane & 16) ? (8 + r) : r;
                hbuf[m * LDSTR + nt * 16 + n] =
                    (unsigned short)(vv[r >> 1] >> ((r & 1) * 16));
            }
        }
    }
    __syncthreads();

    const int mrow = lane & 15;
    const int kb   = (lane & 16) ? 8 : 0;
    const uint4* wg = (const uint4*)Wp + (size_t)(w * 4) * 64 + lane * 2;
    const unsigned short* arow = &hbuf[mrow * LDSTR + kb];

    FragC acc[4];
#pragma unroll
    for (int q = 0; q < 4; ++q)
#pragma unroll
        for (int r = 0; r < 8; ++r) acc[q].x[r] = 0.0f;

#pragma unroll
    for (int kt = 0; kt < KT; ++kt) {
        FragB a;
        const uint4* ap = (const uint4*)(arow + kt * 32);
        uint4 a0 = ap[0], a1 = ap[2];
        a.w[0] = a0.x; a.w[1] = a0.y; a.w[2] = a0.z; a.w[3] = a0.w;
        a.w[4] = a1.x; a.w[5] = a1.y; a.w[6] = a1.z; a.w[7] = a1.w;
#pragma unroll
        for (int q = 0; q < 4; ++q) {
            FragB b = ld_frag_g(wg + ((size_t)kt * NT + q) * 64);
            acc[q].f = __builtin_amdgcn_wmma_f32_16x16x32_bf16(
                false, a.bf, false, b.bf, (short)0, acc[q].f, false, false);
        }
    }

#pragma unroll
    for (int q = 0; q < 4; ++q) {
        int j = (w * 4 + q) * 16 + mrow;
        float bias = bi1[j] + bi2[j];
        float4* d4 = (float4*)(proj +
            (((size_t)t * NT + (w * 4 + q)) * 4 + slice) * 256 + lane * 8);
        d4[0] = make_float4(acc[q].x[0] + bias, acc[q].x[1] + bias,
                            acc[q].x[2] + bias, acc[q].x[3] + bias);
        d4[1] = make_float4(acc[q].x[4] + bias, acc[q].x[5] + bias,
                            acc[q].x[6] + bias, acc[q].x[7] + bias);
    }
}

// ---------------------------------------------------------------------------
// Recurrence: one block per 16-row batch slice; 512 serial steps, no global
// sync. h state in LDS (bf16, padded rows, two barriers/step). W_hh k-tiles
// 0..KT_LDS-1 resident in LDS (chunk-split layout: conflict-free b128);
// k-tiles KT_LDS..15 streamed from L2 with register double-buffering. Proj
// fragments loaded at step start, consumed after the last WMMA.
// ---------------------------------------------------------------------------
template <int STORE_SEQ, int FINAL>
__global__ void __launch_bounds__(256) recur_kernel(
    const float* __restrict__ proj,
    const unsigned int* __restrict__ Whp,
    unsigned int* __restrict__ hseq,
    float* __restrict__ hfin)
{
    __shared__ unsigned int   wlds[KT_LDS * NT * 256];        // 256 KB
    __shared__ unsigned short hbs[16 * (HID + LDS_PAD)];      // 16.6 KB
    constexpr int LDSTR = HID + LDS_PAD;
    const int slice = blockIdx.x;
    const int tid   = threadIdx.x;
    const int w     = tid >> 5;
    const int lane  = tid & 31;
    const int mrow  = lane & 15;
    const int kb    = (lane & 16) ? 8 : 0;
    const int n     = lane & 15;

    // Preload W_hh k-tiles 0..KT_LDS-1 into LDS, chunk-split per lane:
    // chunk half h of lane L -> frag*64 + h*32 + L   (uint4 units)
    {
        const uint4* src = (const uint4*)Whp;
        uint4* dst = (uint4*)wlds;
        const int nch = KT_LDS * NT * 64;
        for (int c = tid; c < nch; c += 256) {
            int frag = c >> 6, ch = c & 63;
            dst[(size_t)frag * 64 + (ch & 1) * 32 + (ch >> 1)] = src[c];
        }
    }
    for (int i = tid; i < 16 * LDSTR; i += 256) hbs[i] = 0;   // h0 = 0
    __syncthreads();

    // Per-wave base pointers (advanced by constants per step).
    const float4* pptr = (const float4*)proj +
                         ((size_t)(w * 4) * 4 + slice) * 64 + lane * 2;
    uint4* sptr = (uint4*)hseq + ((size_t)(w * 4) * 4 + slice) * 32 + lane;
    const uint4* wg  = (const uint4*)Whp + (size_t)(w * 4) * 64 + lane * 2;
    const uint4* wl  = (const uint4*)wlds + (size_t)(w * 4) * 64 + lane;
    const unsigned short* arow = &hbs[mrow * LDSTR + kb];

    float hv[4][8];
    for (int t = 0; t < T_SZ; ++t) {
        // 1. Issue proj fragment loads (consumed at step end).
        float4 pv[4][2];
#pragma unroll
        for (int q = 0; q < 4; ++q) {
            pv[q][0] = pptr[q * 256];
            pv[q][1] = pptr[q * 256 + 1];
        }
        // 2. Prefetch first streamed W k-tile into registers.
        FragB bb[2][4];
#pragma unroll
        for (int q = 0; q < 4; ++q)
            bb[0][q] = ld_frag_g(wg + ((size_t)KT_LDS * NT + q) * 64);

        FragC acc[4];
#pragma unroll
        for (int q = 0; q < 4; ++q)
#pragma unroll
            for (int r = 0; r < 8; ++r) acc[q].x[r] = 0.0f;

        // 3. LDS-resident k-tiles.
#pragma unroll
        for (int kt = 0; kt < KT_LDS; ++kt) {
            FragB a;
            const uint4* ap = (const uint4*)(arow + kt * 32);
            uint4 a0 = ap[0], a1 = ap[2];
            a.w[0] = a0.x; a.w[1] = a0.y; a.w[2] = a0.z; a.w[3] = a0.w;
            a.w[4] = a1.x; a.w[5] = a1.y; a.w[6] = a1.z; a.w[7] = a1.w;
#pragma unroll
            for (int q = 0; q < 4; ++q) {
                const uint4* bp = wl + ((size_t)kt * NT + q) * 64;
                FragB b;
                uint4 b0 = bp[0], b1 = bp[32];
                b.w[0] = b0.x; b.w[1] = b0.y; b.w[2] = b0.z; b.w[3] = b0.w;
                b.w[4] = b1.x; b.w[5] = b1.y; b.w[6] = b1.z; b.w[7] = b1.w;
                acc[q].f = __builtin_amdgcn_wmma_f32_16x16x32_bf16(
                    false, a.bf, false, b.bf, (short)0, acc[q].f, false, false);
            }
        }
        // 4. Streamed k-tiles with register double-buffering.
#pragma unroll
        for (int kt = KT_LDS; kt < KT_ALL; ++kt) {
            const int cb = kt & 1;
            if (kt + 1 < KT_ALL) {
#pragma unroll
                for (int q = 0; q < 4; ++q)
                    bb[cb ^ 1][q] =
                        ld_frag_g(wg + ((size_t)(kt + 1) * NT + q) * 64);
            }
            FragB a;
            const uint4* ap = (const uint4*)(arow + kt * 32);
            uint4 a0 = ap[0], a1 = ap[2];
            a.w[0] = a0.x; a.w[1] = a0.y; a.w[2] = a0.z; a.w[3] = a0.w;
            a.w[4] = a1.x; a.w[5] = a1.y; a.w[6] = a1.z; a.w[7] = a1.w;
#pragma unroll
            for (int q = 0; q < 4; ++q)
                acc[q].f = __builtin_amdgcn_wmma_f32_16x16x32_bf16(
                    false, a.bf, false, bb[cb][q].bf, (short)0, acc[q].f,
                    false, false);
        }
        __syncthreads();   // all h reads done

        // 5. tanh(acc + proj), write h' (LDS rows + optional seq fragments).
#pragma unroll
        for (int q = 0; q < 4; ++q) {
            const float* pf = (const float*)&pv[q][0];
#pragma unroll
            for (int r = 0; r < 8; ++r) hv[q][r] = tanhf(acc[q].x[r] + pf[r]);
            unsigned int pk[4];
#pragma unroll
            for (int p = 0; p < 4; ++p)
                pk[p] = (unsigned int)f2bf(hv[q][2 * p]) |
                        ((unsigned int)f2bf(hv[q][2 * p + 1]) << 16);
#pragma unroll
            for (int r = 0; r < 8; ++r) {
                int m = (lane & 16) ? (8 + r) : r;
                hbs[m * LDSTR + (w * 4 + q) * 16 + n] =
                    (unsigned short)(pk[r >> 1] >> ((r & 1) * 16));
            }
            if (STORE_SEQ)
                sptr[q * 128] = make_uint4(pk[0], pk[1], pk[2], pk[3]);
        }
        __syncthreads();   // h' visible before next step's reads
        pptr += 8192;      // 32*4*256 floats = 8192 float4 per t
        if (STORE_SEQ) sptr += 4096;   // 32*4*128 uints = 4096 uint4 per t
    }

    if (FINAL) {
#pragma unroll
        for (int q = 0; q < 4; ++q)
#pragma unroll
            for (int r = 0; r < 8; ++r) {
                int m = (lane & 16) ? (8 + r) : r;
                hfin[(size_t)(slice * 16 + m) * HID + (w * 4 + q) * 16 + n] =
                    hv[q][r];
            }
    }
}

// ---------------------------------------------------------------------------
// Classifier: out[b][c] = h_final[b] . Wc[c] + bc[c]   (64x2, fp32 VALU)
// ---------------------------------------------------------------------------
__global__ void classify_kernel(const float* __restrict__ hfin,
                                const float* __restrict__ Wc,
                                const float* __restrict__ bc,
                                float* __restrict__ out)
{
    int tid = threadIdx.x;
    if (tid >= B_SZ * 2) return;
    int b = tid >> 1, c = tid & 1;
    float s = bc[c];
    const float* hr = hfin + (size_t)b * HID;
    const float* wr = Wc + (size_t)c * HID;
    for (int j = 0; j < HID; ++j) s += hr[j] * wr[j];
    out[b * 2 + c] = s;
}

extern "C" void kernel_launch(void* const* d_in, const int* in_sizes, int n_in,
                              void* d_out, int out_size, void* d_ws, size_t ws_size,
                              hipStream_t stream) {
    (void)in_sizes; (void)n_in; (void)out_size; (void)ws_size;

    const float* x     = (const float*)d_in[0];
    const float* W_ih0 = (const float*)d_in[1];
    const float* W_hh0 = (const float*)d_in[2];
    const float* b_ih0 = (const float*)d_in[3];
    const float* b_hh0 = (const float*)d_in[4];
    const float* W_ih1 = (const float*)d_in[5];
    const float* W_hh1 = (const float*)d_in[6];
    const float* b_ih1 = (const float*)d_in[7];
    const float* b_hh1 = (const float*)d_in[8];
    const float* Wc    = (const float*)d_in[9];
    const float* bc    = (const float*)d_in[10];

    // Workspace carve (~98 MB total)
    char* ws = (char*)d_ws;
    unsigned int* wih0p = (unsigned int*)ws; ws += (size_t)8  * 32 * 1024;  // 256 KB
    unsigned int* whh0p = (unsigned int*)ws; ws += (size_t)16 * 32 * 1024;  // 512 KB
    unsigned int* wih1p = (unsigned int*)ws; ws += (size_t)16 * 32 * 1024;  // 512 KB
    unsigned int* whh1p = (unsigned int*)ws; ws += (size_t)16 * 32 * 1024;  // 512 KB
    float*        proj  = (float*)ws;        ws += (size_t)T_SZ * 32 * 4 * 1024; // 64 MB
    unsigned int* hseq  = (unsigned int*)ws; ws += (size_t)T_SZ * 32 * 4 * 512;  // 32 MB
    float*        hfin  = (float*)ws;        ws += (size_t)B_SZ * HID * 4;       // 128 KB

    // Pre-pack all weights into WMMA B-fragment bf16 layout.
    pack_w_kernel<<<(8  * 32 * 32 + 255) / 256, 256, 0, stream>>>(W_ih0, wih0p, IN0);
    pack_w_kernel<<<(16 * 32 * 32 + 255) / 256, 256, 0, stream>>>(W_hh0, whh0p, HID);
    pack_w_kernel<<<(16 * 32 * 32 + 255) / 256, 256, 0, stream>>>(W_ih1, wih1p, HID);
    pack_w_kernel<<<(16 * 32 * 32 + 255) / 256, 256, 0, stream>>>(W_hh1, whh1p, HID);

    // Layer 0: input projection, then batch-sliced recurrence (emits h-seq frags).
    proj_kernel<IN0, 0><<<T_SZ * 4, 256, 0, stream>>>(x, wih0p, b_ih0, b_hh0, proj);
    recur_kernel<1, 0><<<4, 256, 0, stream>>>(proj, whh0p, hseq, nullptr);

    // Layer 1: projection from bf16 h-seq fragments, then recurrence -> final h.
    proj_kernel<HID, 1><<<T_SZ * 4, 256, 0, stream>>>(hseq, wih1p, b_ih1, b_hh1, proj);
    recur_kernel<0, 1><<<4, 256, 0, stream>>>(proj, whh1p, nullptr, hfin);

    // Classifier on last timestep.
    classify_kernel<<<1, 128, 0, stream>>>(hfin, Wc, bc, (float*)d_out);
}
